// _FakePackedExperts_9302899163574
// MI455X (gfx1250) — compile-verified, matrix operand order checked by reading, not run on hardware
//
#include <hip/hip_runtime.h>

// Problem constants (from reference)
#define T_TOK 1024
#define H_DIM 1024
#define I_DIM 512
#define E_NUM 16
#define TOPK  4
#define NROWS (T_TOK * TOPK)   // 4096 routed rows total

// Tile config: 256 threads = 8 wave32s arranged 2(M) x 4(N)
#define BM 32
#define BN 128
#define BK 32
#define KSTR 40                 // BK + 8 halves pad: 80B row stride (16B aligned)

// Workspace layout (bytes)
#define WS_SEG_BASE 0                        // 16 * i32
#define WS_SEG_CNT  64                       // 16 * i32
#define WS_ROWS     256                      // 4096 * i32
#define WS_WTS      16640                    // 4096 * f32
#define WS_H        33024                    // (4096+32) * 512 * f16  (~4.1 MB)

typedef _Float16 v16h __attribute__((ext_vector_type(16)));
typedef _Float16 v8h  __attribute__((ext_vector_type(8)));
typedef _Float16 v4h  __attribute__((ext_vector_type(4)));
typedef float    v8f  __attribute__((ext_vector_type(8)));
typedef float    v4f  __attribute__((ext_vector_type(4)));

__device__ __forceinline__ v4h cvt4(v4f f) {
  v4h r;
  r.x = (_Float16)f.x; r.y = (_Float16)f.y;
  r.z = (_Float16)f.z; r.w = (_Float16)f.w;
  return r;
}

__device__ __forceinline__ v16h frag_ld(const _Float16* p) {
  // Two 16B LDS chunks -> ds_load_b128 x2; p must be 16B aligned.
  v16h r;
  ((v8h*)&r)[0] = *(const v8h*)(p);
  ((v8h*)&r)[1] = *(const v8h*)(p + 16);
  return r;
}

// ---------------------------------------------------------------------------
__global__ void moe_zero_out(float* __restrict__ out) {
  int i = blockIdx.x * blockDim.x + threadIdx.x;
  v4f z = {};
  ((v4f*)out)[i] = z;
}

// ---------------------------------------------------------------------------
// Routing: counts -> exclusive prefix -> packed (row, weight) lists per expert.
__global__ void moe_route(const int* __restrict__ topk_idx,
                          const float* __restrict__ topk_w,
                          int* __restrict__ seg_base, int* __restrict__ seg_cnt,
                          int* __restrict__ rows, float* __restrict__ wts) {
  __shared__ int cnt[E_NUM];
  __shared__ int cur[E_NUM];
  const int tid = threadIdx.x;
  if (tid < E_NUM) cnt[tid] = 0;
  __syncthreads();
  for (int i = tid; i < NROWS; i += blockDim.x)
    atomicAdd(&cnt[topk_idx[i]], 1);
  __syncthreads();
  if (tid == 0) {
    int run = 0;
    for (int e = 0; e < E_NUM; ++e) {
      seg_base[e] = run;
      cur[e] = run;
      seg_cnt[e] = cnt[e];
      run += cnt[e];
    }
  }
  __syncthreads();
  for (int i = tid; i < NROWS; i += blockDim.x) {
    int e = topk_idx[i];
    int p = atomicAdd(&cur[e], 1);
    rows[p] = i / TOPK;
    wts[p]  = topk_w[i];
  }
}

// ---------------------------------------------------------------------------
// GEMM1 (fused): h = silu(x @ Wg^T) * (x @ Wu^T), gathered rows, f16 output.
// Software-pipelined: global loads for tile k+1 overlap WMMA on tile k,
// double-buffered LDS, one barrier per K-step.
__global__ __launch_bounds__(256) void moe_gemm1(
    const float* __restrict__ x,            // [T, H]
    const float* __restrict__ wgu,          // [E, 2I, H]
    const int* __restrict__ seg_base, const int* __restrict__ seg_cnt,
    const int* __restrict__ rows,
    _Float16* __restrict__ hbuf) {          // [NROWS+32, I]
  const int e = blockIdx.z;
  const int cnt = seg_cnt[e];
  const int m0 = blockIdx.y * BM;
  if (m0 >= cnt) return;
  const int base = seg_base[e];
  const int n0 = blockIdx.x * BN;           // column within [0, I)

  __shared__ __align__(16) _Float16 As[2][BM * KSTR];
  __shared__ __align__(16) _Float16 Bg[2][BN * KSTR];
  __shared__ __align__(16) _Float16 Bu[2][BN * KSTR];

  const int tid  = threadIdx.x;
  const int lane = tid & 31;
  const int wave = tid >> 5;
  const int m_woff = (wave & 1) * 16;       // 2 waves in M
  const int n_woff = (wave >> 1) * 32;      // 4 waves in N

  // Staging-load mapping: 8 threads per row, float4 each.
  const int lrow = tid >> 3;                // 0..31
  const int lseg = tid & 7;                 // 0..7
  const int gm = m0 + lrow;
  const int grow = (gm < cnt) ? rows[base + gm] : 0;
  const float* aptr = x + (size_t)grow * H_DIM + lseg * 4;

  const float* wg = wgu + (size_t)e * 2 * I_DIM * H_DIM;
  const float* wu = wg + (size_t)I_DIM * H_DIM;
  const float* bgp = wg + (size_t)(n0 + lrow) * H_DIM + lseg * 4;
  const float* bup = wu + (size_t)(n0 + lrow) * H_DIM + lseg * 4;
  const int st_off = lrow * KSTR + lseg * 4;   // LDS store offset (halves)
#define BROWSTEP (32 * H_DIM)                   // 32 weight rows per pass

  // Fragment LDS offsets (halves): lane l holds K={8*lh..}, {16+8*lh..}
  const int lm = lane & 15;
  const int lh = lane >> 4;
  const int a_off  = (m_woff + lm) * KSTR + lh * 8;
  const int b0_off = (n_woff + lm) * KSTR + lh * 8;
  const int b1_off = (n_woff + 16 + lm) * KSTR + lh * 8;

  v8f accg0 = {}, accg1 = {}, accu0 = {}, accu1 = {};

  // --- prologue: stage tile k0=0 into buffer 0
  {
    *(v4h*)&As[0][st_off] = cvt4(*(const v4f*)(aptr));
#pragma unroll
    for (int r = 0; r < 4; ++r) {
      *(v4h*)&Bg[0][r * 32 * KSTR + st_off] = cvt4(*(const v4f*)(bgp + (size_t)r * BROWSTEP));
      *(v4h*)&Bu[0][r * 32 * KSTR + st_off] = cvt4(*(const v4f*)(bup + (size_t)r * BROWSTEP));
    }
  }
  __syncthreads();

  int cur = 0;
  for (int k0 = 0; k0 < H_DIM; k0 += BK) {
    const int k1 = k0 + BK;
    const bool more = (k1 < H_DIM);

    // Issue next tile's global loads (non-blocking w.r.t. the WMMA below).
    v4f na = {}, ng[4], nu[4];
    if (more) {
      na = *(const v4f*)(aptr + k1);
#pragma unroll
      for (int r = 0; r < 4; ++r) {
        ng[r] = *(const v4f*)(bgp + (size_t)r * BROWSTEP + k1);
        nu[r] = *(const v4f*)(bup + (size_t)r * BROWSTEP + k1);
      }
      // speculative prefetch two tiles ahead (dropped silently if OOB)
      __builtin_prefetch(bgp + 2 * BK + k0, 0, 0);
      __builtin_prefetch(bup + 2 * BK + k0, 0, 0);
    }

    // Matrix math on current LDS buffer while the loads are in flight.
    v16h a   = frag_ld(&As[cur][a_off]);
    v16h bg0 = frag_ld(&Bg[cur][b0_off]);
    v16h bg1 = frag_ld(&Bg[cur][b1_off]);
    v16h bu0 = frag_ld(&Bu[cur][b0_off]);
    v16h bu1 = frag_ld(&Bu[cur][b1_off]);
    accg0 = __builtin_amdgcn_wmma_f32_16x16x32_f16(false, a, false, bg0, (short)0, accg0, false, false);
    accg1 = __builtin_amdgcn_wmma_f32_16x16x32_f16(false, a, false, bg1, (short)0, accg1, false, false);
    accu0 = __builtin_amdgcn_wmma_f32_16x16x32_f16(false, a, false, bu0, (short)0, accu0, false, false);
    accu1 = __builtin_amdgcn_wmma_f32_16x16x32_f16(false, a, false, bu1, (short)0, accu1, false, false);

    // Convert + stage next tile into the other buffer.
    if (more) {
      const int nxt = cur ^ 1;
      *(v4h*)&As[nxt][st_off] = cvt4(na);
#pragma unroll
      for (int r = 0; r < 4; ++r) {
        *(v4h*)&Bg[nxt][r * 32 * KSTR + st_off] = cvt4(ng[r]);
        *(v4h*)&Bu[nxt][r * 32 * KSTR + st_off] = cvt4(nu[r]);
      }
    }
    __syncthreads();
    cur ^= 1;
  }

  // Epilogue: silu(gate) * up -> f16 h buffer (guard padded rows).
#pragma unroll
  for (int sub = 0; sub < 2; ++sub) {
    v8f g = sub ? accg1 : accg0;
    v8f u = sub ? accu1 : accu0;
    int ncol = n0 + n_woff + sub * 16 + lm;
#pragma unroll
    for (int v = 0; v < 8; ++v) {
      int m = m_woff + lh * 8 + v;
      if (m0 + m < cnt) {
        float gv = g[v];
        float hv = gv / (1.0f + __expf(-gv)) * u[v];
        hbuf[(size_t)(base + m0 + m) * I_DIM + ncol] = (_Float16)hv;
      }
    }
  }
}

// ---------------------------------------------------------------------------
// GEMM2: out[token, :] += w * (h @ Wd^T)   (scatter via atomics)
// Same pipelined structure as GEMM1.
__global__ __launch_bounds__(256) void moe_gemm2(
    const _Float16* __restrict__ hbuf,      // [NROWS+32, I]
    const float* __restrict__ wd,           // [E, H, I]
    const int* __restrict__ seg_base, const int* __restrict__ seg_cnt,
    const int* __restrict__ rows, const float* __restrict__ wts,
    float* __restrict__ out) {              // [T, H]
  const int e = blockIdx.z;
  const int cnt = seg_cnt[e];
  const int m0 = blockIdx.y * BM;
  if (m0 >= cnt) return;
  const int base = seg_base[e];
  const int n0 = blockIdx.x * BN;           // column within [0, H)

  __shared__ __align__(16) _Float16 As[2][BM * KSTR];
  __shared__ __align__(16) _Float16 Bs[2][BN * KSTR];

  const int tid  = threadIdx.x;
  const int lane = tid & 31;
  const int wave = tid >> 5;
  const int m_woff = (wave & 1) * 16;
  const int n_woff = (wave >> 1) * 32;

  const int lrow = tid >> 3;
  const int lseg = tid & 7;
  const _Float16* aptr = hbuf + (size_t)(base + m0 + lrow) * I_DIM + lseg * 4;
  const float* wde = wd + (size_t)e * H_DIM * I_DIM;
  const float* bp = wde + (size_t)(n0 + lrow) * I_DIM + lseg * 4;
  const int st_off = lrow * KSTR + lseg * 4;
#define BROWSTEP2 (32 * I_DIM)

  const int lm = lane & 15;
  const int lh = lane >> 4;
  const int a_off  = (m_woff + lm) * KSTR + lh * 8;
  const int b0_off = (n_woff + lm) * KSTR + lh * 8;
  const int b1_off = (n_woff + 16 + lm) * KSTR + lh * 8;

  v8f acc0 = {}, acc1 = {};

  // --- prologue: stage tile k0=0 into buffer 0
  {
    *(v4h*)&As[0][st_off] = *(const v4h*)(aptr);
#pragma unroll
    for (int r = 0; r < 4; ++r)
      *(v4h*)&Bs[0][r * 32 * KSTR + st_off] = cvt4(*(const v4f*)(bp + (size_t)r * BROWSTEP2));
  }
  __syncthreads();

  int cur = 0;
  for (int k0 = 0; k0 < I_DIM; k0 += BK) {
    const int k1 = k0 + BK;
    const bool more = (k1 < I_DIM);

    v4h nah = {};
    v4f nb[4];
    if (more) {
      nah = *(const v4h*)(aptr + k1);
#pragma unroll
      for (int r = 0; r < 4; ++r)
        nb[r] = *(const v4f*)(bp + (size_t)r * BROWSTEP2 + k1);
      __builtin_prefetch(bp + 2 * BK + k0, 0, 0);
    }

    v16h a  = frag_ld(&As[cur][a_off]);
    v16h b0 = frag_ld(&Bs[cur][b0_off]);
    v16h b1 = frag_ld(&Bs[cur][b1_off]);
    acc0 = __builtin_amdgcn_wmma_f32_16x16x32_f16(false, a, false, b0, (short)0, acc0, false, false);
    acc1 = __builtin_amdgcn_wmma_f32_16x16x32_f16(false, a, false, b1, (short)0, acc1, false, false);

    if (more) {
      const int nxt = cur ^ 1;
      *(v4h*)&As[nxt][st_off] = nah;
#pragma unroll
      for (int r = 0; r < 4; ++r)
        *(v4h*)&Bs[nxt][r * 32 * KSTR + st_off] = cvt4(nb[r]);
    }
    __syncthreads();
    cur ^= 1;
  }

  // Scale by routing weight and scatter-add into the token's output row.
#pragma unroll
  for (int v = 0; v < 8; ++v) {
    int m = m_woff + lh * 8 + v;
    if (m0 + m < cnt) {
      int idx = base + m0 + m;
      int tok = rows[idx];
      float w = wts[idx];
      float* orow = out + (size_t)tok * H_DIM + n0 + n_woff + lm;
      atomicAdd(orow,      acc0[v] * w);
      atomicAdd(orow + 16, acc1[v] * w);
    }
  }
}

// ---------------------------------------------------------------------------
extern "C" void kernel_launch(void* const* d_in, const int* in_sizes, int n_in,
                              void* d_out, int out_size, void* d_ws, size_t ws_size,
                              hipStream_t stream) {
  const float* x   = (const float*)d_in[0];   // hidden_states [T, H]
  const int*   tki = (const int*)d_in[1];     // top_k_index   [T, K]
  const float* tkw = (const float*)d_in[2];   // top_k_weights [T, K]
  const float* wgu = (const float*)d_in[3];   // gate_up_proj  [E, 2I, H]
  const float* wd  = (const float*)d_in[4];   // down_proj     [E, H, I]
  float* out = (float*)d_out;

  char* ws = (char*)d_ws;
  int*      seg_base = (int*)(ws + WS_SEG_BASE);
  int*      seg_cnt  = (int*)(ws + WS_SEG_CNT);
  int*      rowsp    = (int*)(ws + WS_ROWS);
  float*    wtsp     = (float*)(ws + WS_WTS);
  _Float16* hbuf     = (_Float16*)(ws + WS_H);

  moe_zero_out<<<(T_TOK * H_DIM) / (256 * 4), 256, 0, stream>>>(out);
  moe_route<<<1, 256, 0, stream>>>(tki, tkw, seg_base, seg_cnt, rowsp, wtsp);

  dim3 g1(I_DIM / BN, NROWS / BM, E_NUM);   // (4, 128, 16); blocks early-exit by count
  moe_gemm1<<<g1, 256, 0, stream>>>(x, wgu, seg_base, seg_cnt, rowsp, hbuf);

  dim3 g2(H_DIM / BN, NROWS / BM, E_NUM);   // (8, 128, 16)
  moe_gemm2<<<g2, 256, 0, stream>>>(hbuf, wd, seg_base, seg_cnt, rowsp, wtsp, out);
}
